// EnergyFunction_78529182040170
// MI455X (gfx1250) — matrix operand order repaired and checked by reading, not compiled, not run on hardware
//
#include <hip/hip_runtime.h>
#include <math.h>

typedef __attribute__((ext_vector_type(16))) __bf16 v16bf;
typedef __attribute__((ext_vector_type(8)))  float  v8f;
typedef __attribute__((ext_vector_type(4)))  unsigned int v4u;
typedef __attribute__((ext_vector_type(8)))  int v8i_t;
typedef __attribute__((ext_vector_type(4)))  int v4i_t;

#define EPSF  1e-5f
#define MAXN  (1.0f - 1e-5f)
#define MINN  1e-15f

__device__ __forceinline__ float wave_sum32(float v) {
#pragma unroll
  for (int m = 1; m < 32; m <<= 1) v += __shfl_xor(v, m, 32);
  return v;
}
__device__ __forceinline__ float pair_sum(float v) {
  return v + __shfl_xor(v, 16, 32);
}

// ---- Tensor Data Mover: DMA one 128x128 bf16 matrix Global -> LDS ----
// D# built per CDNA5 ISA ch.8: group0 = {flags, lds_addr, global_addr, type=2},
// group1 = {data_size=2B, tensor_dim0/1=128, tile_dim0/1=128, stride0=128}.
__device__ __forceinline__ void tdm_load_w128(const __bf16* gsrc, __bf16* ldst) {
  unsigned long long ga = (unsigned long long)(uintptr_t)gsrc;
  unsigned int lds = (unsigned int)(uintptr_t)ldst;   // low 32 bits = LDS offset
  v4u g0;
  g0[0] = 1u;                                          // count=1, user descriptor
  g0[1] = lds;                                         // lds_addr [63:32]
  g0[2] = (unsigned int)(ga & 0xFFFFFFFFu);            // global_addr [95:64]
  g0[3] = (unsigned int)((ga >> 32) & 0x01FFFFFFu)     // global_addr [120:96]
        | (2u << 30);                                  // type=2 ("image")
  v8i_t g1;
  g1[0] = 0x00010000;          // workgroup_mask=0, data_size=1 (2 bytes)
  g1[1] = (int)(128u << 16);   // tensor_dim0[15:0]=128 (bits 63:48)
  g1[2] = (int)(128u << 16);   // tensor_dim1[15:0]=128 (bits 95:80)
  g1[3] = (int)(128u << 16);   // tile_dim0=128 (bits 127:112)
  g1[4] = 128;                 // tile_dim1=128, tile_dim2=0
  g1[5] = 128;                 // tensor_dim0_stride=128
  g1[6] = 0;                   // stride hi / dim1_stride (unused, 2D tile)
  g1[7] = 0;
  v4i_t z4 = {0, 0, 0, 0};
#if __has_include(<hip/amd_detail/amd_gfx1250_TDM.h>)
  v8i_t z8 = {0, 0, 0, 0, 0, 0, 0, 0};
  __builtin_amdgcn_tensor_load_to_lds(g0, g1, z4, z4, z8, 0);
#else
  __builtin_amdgcn_tensor_load_to_lds(g0, g1, z4, z4, 0);
#endif
}

// Convert the two 128x128 f32 weight matrices to bf16 (row-major [e][d] kept).
__global__ void prep_w_kernel(const float* __restrict__ w, __bf16* __restrict__ wb, int n) {
  int i = blockIdx.x * blockDim.x + threadIdx.x;
  if (i < n) wb[i] = (__bf16)w[i];
}

// y2[layer] = sum(b^2) per layer.
__global__ void prep_y2_kernel(const float* __restrict__ lb, float* __restrict__ y2out) {
  const int l = threadIdx.x >> 5, ln = threadIdx.x & 31;
  const float* bp = lb + l * 128;
  float s = 0.f;
  for (int c = ln; c < 128; c += 32) { const float t = bp[c]; s += t * t; }
  s = wave_sum32(s);
  if (ln == 0) y2out[l] = s;
}

__global__ __launch_bounds__(128) void hyp_kernel(
    const int*   __restrict__ inp,   // (8192, 51) indices (int32 on device)
    const float* __restrict__ emb,   // (500000, 128)
    const __bf16* __restrict__ wbg,  // (2, 128, 128) bf16, row-major [e][d]
    const float* __restrict__ lb,    // (2, 128)
    const float* __restrict__ y2g,   // (2,) sum(b^2)
    float*       __restrict__ out)   // (8192, 50)
{
  __shared__ __align__(32) float  xbuf[64 * 128];    // 32 KB: x rows -> mx -> x'
  __shared__ __align__(32) __bf16 wbuf0[128 * 128];  // 32 KB: W layer 0 (TDM)
  __shared__ __align__(32) __bf16 wbuf1[128 * 128];  // 32 KB: W layer 1 (TDM)

  const int b   = blockIdx.x;
  const int tid = threadIdx.x;
  const int wv  = tid >> 5;        // wave id: 16-row tile owner
  const int ln  = tid & 31;
  const int m   = ln & 15;         // row-in-tile for WMMA A / epilogue
  const int h   = ln >> 4;         // lane half
  const int rowBase = wv * 16;

  // Wave 0 kicks both W DMAs before the gather hides their latency.
  // readfirstlane makes the predicate wave-uniform -> scalar branch, so
  // waves 1..3 skip the (EXEC-ignoring) tensor op entirely.
  if (__builtin_amdgcn_readfirstlane(tid) == 0) {
    tdm_load_w128(wbg, wbuf0);
    tdm_load_w128(wbg + 128 * 128, wbuf1);
  }

  // ---------------- gather + renormalize + expmap0 ----------------
  float xn_reg = 0.0f;             // _norm(x) for row (rowBase+m), in both lane halves
  for (int r = 0; r < 16; ++r) {
    const int col = rowBase + r;
    float4 e = make_float4(0.f, 0.f, 0.f, 0.f);
    if (col < 51) {
      const int idx = inp[b * 51 + col];
      const float4* p = (const float4*)(emb + (size_t)idx * 128) + ln;
      e = *p;
    }
    float s = e.x * e.x + e.y * e.y + e.z * e.z + e.w * e.w;
    s = wave_sum32(s);                                  // raw sum of squares
    const float n   = sqrtf(fmaxf(s, MINN));
    const float sc1 = (n > MAXN) ? (MAXN / n) : 1.0f;   // renormalize
    const float nn  = fminf(n, MAXN);
    const float sc2 = tanhf(nn) / nn;                   // expmap0
    const float scale = sc1 * sc2;
    const float xn = sqrtf(fmaxf(scale * scale * s, MINN));
    if (r == m) xn_reg = xn;
    float4 xo = make_float4(e.x * scale, e.y * scale, e.z * scale, e.w * scale);
    *(float4*)(xbuf + (rowBase + r) * 128 + ln * 4) = xo;
  }

  // W DMAs must be complete before any wave reads wbuf*.
  if (__builtin_amdgcn_readfirstlane(tid) == 0)
    __builtin_amdgcn_s_wait_tensorcnt(0);
  __syncthreads();

  // ---------------- two mobius_linear layers (no cross-wave deps) ----------------
  for (int layer = 0; layer < 2; ++layer) {
    const __bf16* wl = layer ? wbuf1 : wbuf0;

    // A operand (this wave's 16 rows), contiguous k order (same as B):
    // elems 0..15 -> K = kc*32 + 16h + e
    v16bf A[4];
#pragma unroll
    for (int kc = 0; kc < 4; ++kc) {
      const float* xp = xbuf + (rowBase + m) * 128 + kc * 32 + 16 * h;
#pragma unroll
      for (int e = 0; e < 8; ++e) {
        A[kc][e]     = (__bf16)xp[e];
        A[kc][e + 8] = (__bf16)xp[8 + e];
      }
    }

    // matvec: mx = x * W^T. B layout (32x16): lane holds col n=m, K=kc*32+16h+e.
    // mx overwrites this wave's xbuf rows (A already in registers).
#pragma unroll
    for (int nt = 0; nt < 8; ++nt) {
      v8f acc = {};
#pragma unroll
      for (int kc = 0; kc < 4; ++kc) {
        const v16bf Bv = *(const v16bf*)(wl + (nt * 16 + m) * 128 + kc * 32 + 16 * h);
        acc = __builtin_amdgcn_wmma_f32_16x16x32_bf16(
            false, A[kc], false, Bv, (short)0, acc, false, false);
      }
#pragma unroll
      for (int r = 0; r < 8; ++r)   // D elem r -> row (r + 8h), col nt*16 + m
        xbuf[(rowBase + r + 8 * h) * 128 + nt * 16 + m] = acc[r];
    }

    // epilogue: lane pair (m,h) owns row rowBase+m, cols [64h, 64h+64)
    float4*       mx4 = (float4*)(xbuf + (rowBase + m) * 128 + 64 * h);
    const float4* bv4 = (const float4*)(lb + layer * 128 + 64 * h);
    const float   y2  = y2g[layer];

    float s2s = 0.f, dxy = 0.f;
#pragma unroll 4
    for (int c = 0; c < 16; ++c) {
      const float4 v = mx4[c], w = bv4[c];
      s2s += v.x * v.x + v.y * v.y + v.z * v.z + v.w * v.w;
      dxy += v.x * w.x + v.y * w.y + v.z * w.z + v.w * w.w;
    }
    const float mx2 = pair_sum(s2s);
    const float mxb = pair_sum(dxy);
    const float mxn = sqrtf(fmaxf(mx2, MINN));
    const float r1  = atanhf(fminf(xn_reg, MAXN));
    const float scmv = tanhf(mxn / xn_reg * r1) / mxn;    // mobius_matvec scale
    const float x2 = scmv * scmv * mx2;
    const float xy = scmv * mxb;
    const float numc = (1.f + 2.f * xy + y2) * scmv;      // coeff on mx
    const float numb = 1.f - x2;                          // coeff on b
    const float id = 1.f / fmaxf(1.f + 2.f * xy + x2 * y2, MINN);

    float sa = 0.f;
#pragma unroll 4
    for (int c = 0; c < 16; ++c) {
      const float4 v = mx4[c], w = bv4[c];
      float4 a;
      a.x = (numc * v.x + numb * w.x) * id;
      a.y = (numc * v.y + numb * w.y) * id;
      a.z = (numc * v.z + numb * w.z) * id;
      a.w = (numc * v.w + numb * w.w) * id;
      mx4[c] = a;
      sa += a.x * a.x + a.y * a.y + a.z * a.z + a.w * a.w;
    }
    sa = pair_sum(sa);
    const float an = sqrtf(fmaxf(sa, MINN));
    const float s1 = atanhf(fminf(an, MAXN)) / an;        // logmap0 scale

    float st = 0.f;
    for (int c = 0; c < 16; ++c) {
      const float4 v = mx4[c];
      float4 t;
      t.x = tanhf(s1 * v.x); t.y = tanhf(s1 * v.y);
      t.z = tanhf(s1 * v.z); t.w = tanhf(s1 * v.w);
      mx4[c] = t;
      st += t.x * t.x + t.y * t.y + t.z * t.z + t.w * t.w;
    }
    st = pair_sum(st);
    const float tn  = sqrtf(fmaxf(st, MINN));
    const float se  = tanhf(tn) / tn;                     // expmap0 scale
    const float wn  = se * tn;
    const float sc3 = (wn > MAXN) ? (MAXN / wn) * se : se; // project
#pragma unroll 4
    for (int c = 0; c < 16; ++c) {
      float4 v = mx4[c];
      v.x *= sc3; v.y *= sc3; v.z *= sc3; v.w *= sc3;
      mx4[c] = v;
    }
    xn_reg = sqrtf(fmaxf(sc3 * sc3 * st, MINN));          // norm for next layer
  }

  __syncthreads();
  // ---------------- poincare distance: s = row 0, o = rows 1..50 ----------------
  if (tid < 50) {
    const float4* s4 = (const float4*)xbuf;
    const float4* o4 = (const float4*)(xbuf + (tid + 1) * 128);
    float sq = 0.f, ss = 0.f, oo = 0.f;
    for (int d = 0; d < 32; ++d) {
      const float4 sv = s4[d], ov = o4[d];
      const float dx = sv.x - ov.x, dy = sv.y - ov.y, dz = sv.z - ov.z, dw = sv.w - ov.w;
      sq += dx * dx + dy * dy + dz * dz + dw * dw;
      ss += sv.x * sv.x + sv.y * sv.y + sv.z * sv.z + sv.w * sv.w;
      oo += ov.x * ov.x + ov.y * ov.y + ov.z * ov.z + ov.w * ov.w;
    }
    const float alpha = fmaxf(1.f - ss, EPSF);
    const float beta  = fmaxf(1.f - oo, EPSF);
    float z = 1.f + 2.f * sq / (alpha * beta);
    z = fmaxf(z, 1.f + 1e-7f);
    out[(size_t)b * 50 + tid] = acoshf(z);
  }
}

extern "C" void kernel_launch(void* const* d_in, const int* in_sizes, int n_in,
                              void* d_out, int out_size, void* d_ws, size_t ws_size,
                              hipStream_t stream) {
  (void)in_sizes; (void)n_in; (void)out_size; (void)ws_size;
  const int*   inp = (const int*)d_in[0];
  const float* emb = (const float*)d_in[1];
  const float* lw  = (const float*)d_in[2];
  const float* lbp = (const float*)d_in[3];
  float* out = (float*)d_out;
  __bf16* wbf = (__bf16*)d_ws;                       // 64 KB: bf16 weights
  float*  y2  = (float*)((char*)d_ws + 64 * 1024);   // 8 B: per-layer sum(b^2)

  const int nW = 2 * 128 * 128;
  prep_w_kernel<<<dim3((nW + 255) / 256), dim3(256), 0, stream>>>(lw, wbf, nW);
  prep_y2_kernel<<<dim3(1), dim3(64), 0, stream>>>(lbp, y2);
  hyp_kernel<<<dim3(8192), dim3(128), 0, stream>>>(inp, emb, wbf, lbp, y2, out);
}
